// End2EndClassifierRGCN_63127429316659
// MI455X (gfx1250) — matrix operand order repaired and checked by reading, not compiled, not run on hardware
//
#include <hip/hip_runtime.h>
#include <hip/hip_bf16.h>

// ---------------- problem constants (match reference) ----------------
constexpr int N_   = 100000;  // nodes
constexpr int IN_  = 256;     // input features
constexpr int H_   = 128;     // hidden
constexpr int OUT_ = 16;      // classes
constexpr int R_   = 5;       // relations
constexpr int E_   = 100000;  // edges per relation
// B (bases) = 2, folded into a K=256 concatenated GEMM.

typedef __attribute__((ext_vector_type(16))) __bf16          v16bf;
typedef __attribute__((ext_vector_type(16))) unsigned short  v16u;
typedef __attribute__((ext_vector_type(8)))  float           v8f;

__device__ __forceinline__ unsigned short f2bf(float f) {
    union { float f; unsigned u; } v; v.f = f;
    unsigned u = v.u;
    unsigned r = (u + 0x7FFFu + ((u >> 16) & 1u)) >> 16;   // RNE
    return (unsigned short)r;
}

// ---------------- utility kernels ----------------
__global__ void zero_f32(float* __restrict__ p, long n) {
    long t = (long)blockIdx.x * blockDim.x + threadIdx.x;
    long s = (long)gridDim.x * blockDim.x;
    for (; t < n; t += s) p[t] = 0.0f;
}

// src [K][C] row-major fp32 -> dst bf16 [C][256] at column offset koff
__global__ void transpose_cvt_bf16(const float* __restrict__ src,
                                   unsigned short* __restrict__ dst,
                                   int K, int C, int koff) {
    int t = blockIdx.x * blockDim.x + threadIdx.x;
    if (t >= K * C) return;
    int k = t / C, c = t % C;
    dst[(long)c * 256 + koff + k] = f2bf(src[t]);
}

// count in-degree per (relation, node) into deg[R*N]
__global__ void deg_count(const int* __restrict__ dst_all, float* __restrict__ deg) {
    int t = blockIdx.x * blockDim.x + threadIdx.x;
    if (t >= R_ * E_) return;
    int r = t / E_;
    unsafeAtomicAdd(&deg[(long)r * N_ + dst_all[t]], 1.0f);
}

// deg -> 1/max(deg,1)
__global__ void inv_finalize(float* __restrict__ deg, long n) {
    long t = (long)blockIdx.x * blockDim.x + threadIdx.x;
    if (t < n) deg[t] = 1.0f / fmaxf(deg[t], 1.0f);
}

// ---------------- edge scatter: Z[d][b*128+c] += coef[r][b]*invdeg[d]*h[s][c] ----------------
// coef[r,b] and invdeg[dst] are known before the scatter, so the whole
// basis-combined mean-aggregation collapses into this single atomic pass.
__global__ void scatter_edges(const float* __restrict__ h,
                              const int* __restrict__ src,
                              const int* __restrict__ dst,
                              const float* __restrict__ invdeg,   // pre-offset by r*N
                              const float* __restrict__ coef,     // [R,2]
                              int r,
                              float* __restrict__ Z) {            // [N,256]
    int t = blockIdx.x * blockDim.x + threadIdx.x;
    int e = t >> 5;                       // 32 lanes per edge
    if (e >= E_) return;
    int q = (t & 31) * 4;                 // 4 floats per lane -> 128 per edge
    int s = src[e], d = dst[e];
    float w  = invdeg[d];
    float c0 = coef[r * 2 + 0] * w;
    float c1 = coef[r * 2 + 1] * w;
    const float4 v = *(const float4*)(h + (long)s * 128 + q);
    float* z0 = Z + (long)d * 256 + q;    // basis-0 field
    float* z1 = z0 + 128;                 // basis-1 field
    unsafeAtomicAdd(z0 + 0, c0 * v.x); unsafeAtomicAdd(z0 + 1, c0 * v.y);
    unsafeAtomicAdd(z0 + 2, c0 * v.z); unsafeAtomicAdd(z0 + 3, c0 * v.w);
    unsafeAtomicAdd(z1 + 0, c1 * v.x); unsafeAtomicAdd(z1 + 1, c1 * v.y);
    unsafeAtomicAdd(z1 + 2, c1 * v.z); unsafeAtomicAdd(z1 + 3, c1 * v.w);
}

// ---------------- WMMA GEMM: C[M, NT*16] = A[M,256](fp32->bf16) @ Wt^T + bias ----------------
// Wt is bf16, [NT*16][256] (i.e. column-major W), so B fragments are contiguous.
// One wave owns a 16-row strip and all NT column tiles (A read exactly once).
// A frag (16-bit 16x32): lane<16 holds K 0-15, lane>=16 holds K 16-31, row = lane&15.
// B frag (16-bit 32x16): lane<16 holds K 0-15, lane>=16 holds K 16-31, col = lane&15.
// C frag: VGPR v -> rows v (lanes 0-15) and v+8 (lanes 16-31), col = lane&15.
template <int NT>
__global__ __launch_bounds__(64)
void wmma_gemm_k256(const float* __restrict__ A,
                    const unsigned short* __restrict__ Wt,
                    const float* __restrict__ bias,
                    float* __restrict__ C, int relu) {
    const int lane  = threadIdx.x & 31;
    const int half  = lane >> 4;
    const int l15   = lane & 15;
    const long row0 = ((long)blockIdx.x * 2 + (threadIdx.x >> 5)) * 16;
    const float* arow = A + (row0 + l15) * 256;
    const int abase = half * 16;
    const int bbase = half * 16;

    v8f acc[NT];
#pragma unroll
    for (int n = 0; n < NT; ++n) acc[n] = (v8f){0,0,0,0,0,0,0,0};

    for (int kk = 0; kk < 256; kk += 32) {
        const float4 a0 = *(const float4*)(arow + kk + abase +  0);
        const float4 a1 = *(const float4*)(arow + kk + abase +  4);
        const float4 a2 = *(const float4*)(arow + kk + abase +  8);
        const float4 a3 = *(const float4*)(arow + kk + abase + 12);
        v16u au;
        au[0]=f2bf(a0.x); au[1]=f2bf(a0.y); au[2] =f2bf(a0.z); au[3] =f2bf(a0.w);
        au[4]=f2bf(a1.x); au[5]=f2bf(a1.y); au[6] =f2bf(a1.z); au[7] =f2bf(a1.w);
        au[8]=f2bf(a2.x); au[9]=f2bf(a2.y); au[10]=f2bf(a2.z); au[11]=f2bf(a2.w);
        au[12]=f2bf(a3.x);au[13]=f2bf(a3.y);au[14]=f2bf(a3.z); au[15]=f2bf(a3.w);
        v16bf a = __builtin_bit_cast(v16bf, au);
#pragma unroll
        for (int n = 0; n < NT; ++n) {
            const unsigned short* bp = Wt + ((long)(n * 16 + l15)) * 256 + kk + bbase;
            v16u bu = *(const v16u*)bp;              // 32B contiguous bf16
            v16bf b = __builtin_bit_cast(v16bf, bu);
            acc[n] = __builtin_amdgcn_wmma_f32_16x16x32_bf16(
                false, a, false, b, (short)0, acc[n], false, false);
        }
    }

    const int Nc = NT * 16;
#pragma unroll
    for (int n = 0; n < NT; ++n) {
        int col  = n * 16 + l15;
        float bv = bias[col];
#pragma unroll
        for (int v = 0; v < 8; ++v) {
            long r = row0 + half * 8 + v;
            float val = acc[n][v] + bv;
            if (relu) val = fmaxf(val, 0.0f);
            C[r * Nc + col] = val;
        }
    }
}

// ---------------- host-side orchestration ----------------
extern "C" void kernel_launch(void* const* d_in, const int* in_sizes, int n_in,
                              void* d_out, int out_size, void* d_ws, size_t ws_size,
                              hipStream_t stream) {
    const float* x       = (const float*)d_in[0];
    const int*   src     = (const int*)  d_in[1];
    const int*   dst     = (const int*)  d_in[2];
    const float* w_embed = (const float*)d_in[3];
    const float* b_embed = (const float*)d_in[4];
    const float* basis1  = (const float*)d_in[5];
    const float* coef1   = (const float*)d_in[6];
    const float* bias1   = (const float*)d_in[7];
    const float* basis2  = (const float*)d_in[8];
    const float* coef2   = (const float*)d_in[9];
    const float* bias2   = (const float*)d_in[10];
    float* out = (float*)d_out;

    // workspace carve-up (128B aligned pieces)
    char* w = (char*)d_ws;
    float* h      = (float*)w; w += (size_t)N_ * 128 * 4;           // 51.2 MB
    float* Z      = (float*)w; w += (size_t)N_ * 256 * 4;           // 102.4 MB
    float* invdeg = (float*)w; w += (size_t)R_ * N_ * 4;            // 2.0 MB
    unsigned short* WtE = (unsigned short*)w; w += 128 * 256 * 2;
    unsigned short* Wt1 = (unsigned short*)w; w += 128 * 256 * 2;
    unsigned short* Wt2 = (unsigned short*)w; w += 16  * 256 * 2;

    // --- weights: fp32 -> bf16, transposed to [col][k=256] ---
    transpose_cvt_bf16<<<(256*128+255)/256, 256, 0, stream>>>(w_embed, WtE, 256, 128, 0);
    transpose_cvt_bf16<<<(128*128+255)/256, 256, 0, stream>>>(basis1,            Wt1, 128, 128, 0);
    transpose_cvt_bf16<<<(128*128+255)/256, 256, 0, stream>>>(basis1 + 128*128,  Wt1, 128, 128, 128);
    transpose_cvt_bf16<<<(128*16 +255)/256, 256, 0, stream>>>(basis2,            Wt2, 128, 16,  0);
    transpose_cvt_bf16<<<(128*16 +255)/256, 256, 0, stream>>>(basis2 + 128*16,   Wt2, 128, 16,  128);

    // --- degrees (shared by both layers: same graph) ---
    zero_f32<<<2048, 256, 0, stream>>>(invdeg, (long)R_ * N_);
    deg_count<<<(R_*E_ + 255)/256, 256, 0, stream>>>(dst, invdeg);
    inv_finalize<<<((long)R_*N_ + 255)/256, 256, 0, stream>>>(invdeg, (long)R_ * N_);

    // --- embed: h = x @ w_embed + b_embed (no activation) ---
    wmma_gemm_k256<8><<<N_/32, 64, 0, stream>>>(x, WtE, b_embed, h, 0);

    // --- layer 1: Z = relation/basis-folded mean aggregation of h; h = relu(Z @ [V0;V1] + bias1) ---
    zero_f32<<<4096, 256, 0, stream>>>(Z, (long)N_ * 256);
    for (int r = 0; r < R_; ++r)
        scatter_edges<<<(E_*32 + 255)/256, 256, 0, stream>>>(
            h, src + (long)r*E_, dst + (long)r*E_, invdeg + (long)r*N_, coef1, r, Z);
    wmma_gemm_k256<8><<<N_/32, 64, 0, stream>>>(Z, Wt1, bias1, h, 1);   // overwrite h with h1

    // --- layer 2: same aggregation on h1, project to 16 classes, no activation ---
    zero_f32<<<4096, 256, 0, stream>>>(Z, (long)N_ * 256);
    for (int r = 0; r < R_; ++r)
        scatter_edges<<<(E_*32 + 255)/256, 256, 0, stream>>>(
            h, src + (long)r*E_, dst + (long)r*E_, invdeg + (long)r*N_, coef2, r, Z);
    wmma_gemm_k256<1><<<N_/32, 64, 0, stream>>>(Z, Wt2, bias2, out, 0);
}
